// PolicyRN34_67731634258293
// MI455X (gfx1250) — matrix operand order repaired and checked by reading, not compiled, not run on hardware
//
#include <hip/hip_runtime.h>

typedef __attribute__((ext_vector_type(16))) __bf16 v16bf;
typedef __attribute__((ext_vector_type(8)))  __bf16 v8bf;
typedef __attribute__((ext_vector_type(8)))  float  v8f;

#define MODE_BN       0
#define MODE_BNRELU   1
#define MODE_RES_GATE 2

// exact for 0 <= p < 2^16, 1 <= d < 2^16, M = ceil(2^32/d)
__device__ __forceinline__ int fastdiv(int p, unsigned M) {
  return (int)(((unsigned long long)(unsigned)p * M) >> 32);
}

// ---------------- prep kernels ----------------

__global__ void k_bnfold(const float* g, const float* b, const float* m, const float* v,
                         float* scale, float* shift, int C) {
  int c = blockIdx.x * blockDim.x + threadIdx.x;
  if (c < C) {
    float s = g[c] * rsqrtf(v[c] + 1e-5f);
    scale[c] = s;
    shift[c] = b[c] - m[c] * s;
  }
}

// w: [Cout][Cin][taps] fp32 -> wt: [taps][Cout][Cin] bf16 with the Cin dimension
// kept in natural order inside every 32-channel block:
// lane-half 0 holds K={0..15}, lane-half 1 holds K={16..31}, so each
// lane's 16 A elements are one contiguous 32B load.
__global__ void k_wconv(const float* w, __bf16* wt, int Cout, int Cin, int taps) {
  int i = blockIdx.x * blockDim.x + threadIdx.x;
  int total = Cout * Cin * taps;
  if (i >= total) return;
  int tap = i % taps;
  int ci  = (i / taps) % Cin;
  int co  = i / (taps * Cin);
  int blk = ci >> 5, r = ci & 31;
  int pos = r;
  wt[tap * Cout * Cin + co * Cin + (blk << 5) + pos] = (__bf16)w[i];
}

// ---------------- base conv 7x7 s2 p3 + BN + ReLU (Cin=3, ~3% of FLOPs) ----------------
// Output in NHWC bf16: [32][112][112][64]

__global__ void k_baseconv(const float* __restrict__ x, const float* __restrict__ w,
                           const float* __restrict__ scale, const float* __restrict__ shift,
                           __bf16* __restrict__ out) {
  long long i = (long long)blockIdx.x * blockDim.x + threadIdx.x;
  const long long total = 32LL * 112 * 112 * 64;
  if (i >= total) return;
  int c  = (int)(i % 64);
  int xo = (int)((i / 64) % 112);
  int yo = (int)((i / (64 * 112)) % 112);
  int n  = (int)(i / (64LL * 112 * 112));
  float acc = 0.f;
  for (int ci = 0; ci < 3; ++ci) {
    for (int ky = 0; ky < 7; ++ky) {
      int iy = yo * 2 - 3 + ky;
      if (iy < 0 || iy >= 224) continue;
      const float* xr = x + (((n * 3 + ci) * 224 + iy) * 224);
      const float* wr = w + ((c * 3 + ci) * 7 + ky) * 7;
      for (int kx = 0; kx < 7; ++kx) {
        int ix = xo * 2 - 3 + kx;
        if (ix < 0 || ix >= 224) continue;
        acc += xr[ix] * wr[kx];
      }
    }
  }
  float val = acc * scale[c] + shift[c];
  out[i] = (__bf16)fmaxf(val, 0.f);
}

// ---------------- maxpool 3x3 s2 p1 : 112 -> 56 (NHWC) ----------------

__global__ void k_maxpool(const __bf16* __restrict__ in, __bf16* __restrict__ out) {
  long long i = (long long)blockIdx.x * blockDim.x + threadIdx.x;
  const long long total = 32LL * 56 * 56 * 64;
  if (i >= total) return;
  int c  = (int)(i % 64);
  int xo = (int)((i / 64) % 56);
  int yo = (int)((i / (64 * 56)) % 56);
  int n  = (int)(i / (64LL * 56 * 56));
  float mx = -3.4e38f;
  for (int ky = 0; ky < 3; ++ky) {
    int iy = yo * 2 - 1 + ky;
    if (iy < 0 || iy >= 112) continue;
    for (int kx = 0; kx < 3; ++kx) {
      int ix = xo * 2 - 1 + kx;
      if (ix < 0 || ix >= 112) continue;
      mx = fmaxf(mx, (float)in[((n * 112 + iy) * 112 + ix) * 64 + c]);
    }
  }
  out[i] = (__bf16)mx;
}

// ---------------- implicit-GEMM conv via v_wmma_f32_16x16x32_bf16 (NHWC) ----------------
// One wave per block computing a 32(Cout) x 64(pixel) tile: 2 A-frags x 4 B-frags
// -> 8 WMMAs per K-step. NHWC makes every B-fragment a contiguous 32B load per lane
// (16 consecutive channels); pre-swizzled weights make every A-fragment a single
// contiguous 32B load; the epilogue is a packed 16B store per fragment.
// A-frag: lane row M = lane&15; pre-swizzle puts its 16 elements at [k0 + 16*(lane>=16)].
// B-frag (16-bit B 32x16): lane col N = lane&15; k(e) = 16*(lane>=16) + e  (contiguous)
// D: acc[r] -> M = r + 8*(lane>=16), N = lane&15
__global__ __launch_bounds__(32)
void k_conv_wmma(const __bf16* __restrict__ X, const __bf16* __restrict__ Wt,
                 const float* __restrict__ scale, const float* __restrict__ shift,
                 __bf16* __restrict__ Out, const int* __restrict__ policy,
                 int polBlock, int mode,
                 int Cin, int Cout, int Hin, int Win, int Hout, int Wout,
                 int stride, int pad, int taps, unsigned Mw) {
  const int lane = threadIdx.x;
  const int half = lane >> 4;
  const int l15  = lane & 15;
  const int n     = blockIdx.z;
  const int cout0 = blockIdx.y * 32;
  const int p0    = blockIdx.x * 64;
  const int HWo = Hout * Wout;

  // per-fragment pixel geometry (4 pixel fragments of 16)
  int py[4], px[4];
  bool pvv[4];
#pragma unroll
  for (int f = 0; f < 4; ++f) {
    const int p = p0 + f * 16 + l15;
    pvv[f] = p < HWo;
    const int oy = fastdiv(p, Mw);
    py[f] = oy * stride - pad;
    px[f] = (p - oy * Wout) * stride - pad;
  }
  const int nbase = n * Hin * Win;            // pixel units
  const int kfrag = half * 16;
  const __bf16* arow0 = Wt + (cout0 + l15) * Cin + kfrag;
  const __bf16* arow1 = arow0 + 16 * Cin;

  v8f acc[2][4] = {};

  for (int tap = 0; tap < taps; ++tap) {
    const int offy = (taps == 9) ? (tap / 3) : 0;
    const int offx = (taps == 9) ? (tap % 3) : 0;
    const int woff = tap * Cout * Cin;
    // per-fragment tap base address + validity (zero-padding)
    const __bf16* bptr[4];
    bool bv[4];
#pragma unroll
    for (int f = 0; f < 4; ++f) {
      const int iy = py[f] + offy, ix = px[f] + offx;
      bv[f] = pvv[f] && (unsigned)iy < (unsigned)Hin && (unsigned)ix < (unsigned)Win;
      bptr[f] = X + (nbase + iy * Win + ix) * Cin + kfrag;
    }

    for (int k0 = 0; k0 < Cin; k0 += 32) {
      if (k0 + 32 < Cin) __builtin_prefetch(arow0 + woff + k0 + 32, 0, 3);  // global_prefetch_b8

      // two A fragments: single contiguous 32B load each (pre-swizzled weights)
      v16bf a0 = *(const v16bf*)(const void*)(arow0 + woff + k0);
      v16bf a1 = *(const v16bf*)(const void*)(arow1 + woff + k0);

      // four B fragments: one contiguous 32B load per lane each; 8 WMMAs
#pragma unroll
      for (int f = 0; f < 4; ++f) {
        v16bf b = {};
        if (bv[f]) b = *(const v16bf*)(const void*)(bptr[f] + k0);
        acc[0][f] = __builtin_amdgcn_wmma_f32_16x16x32_bf16(false, a0, false, b, (short)0,
                                                            acc[0][f], false, false);
        acc[1][f] = __builtin_amdgcn_wmma_f32_16x16x32_bf16(false, a1, false, b, (short)0,
                                                            acc[1][f], false, false);
      }
    }
  }

  // ---- epilogue: BN (+ReLU) (+residual & per-sample gate, in place), packed 16B stores ----
  const bool keep = (mode == MODE_RES_GATE) ? (policy[polBlock * 64 + n] != 0) : true;
#pragma unroll
  for (int f = 0; f < 4; ++f) {
    const int p = p0 + f * 16 + l15;
    if (p >= HWo) continue;
    const int obase = (n * HWo + p) * Cout;      // NHWC
#pragma unroll
    for (int g = 0; g < 2; ++g) {
      const int c0 = cout0 + g * 16 + half * 8;  // 8 consecutive channels
      __bf16* optr = Out + obase + c0;
      v8bf ov;
      if (mode == MODE_RES_GATE) {
        v8bf rv = *(const v8bf*)(const void*)optr;   // residual, 1:1 in-place
#pragma unroll
        for (int r = 0; r < 8; ++r) {
          const float val = acc[g][f][r] * scale[c0 + r] + shift[c0 + r];
          const float res = (float)rv[r];
          const float fx  = fmaxf(res + val, 0.f);
          ov[r] = (__bf16)(keep ? fx : res);
        }
      } else {
#pragma unroll
        for (int r = 0; r < 8; ++r) {
          float val = acc[g][f][r] * scale[c0 + r] + shift[c0 + r];
          if (mode == MODE_BNRELU) val = fmaxf(val, 0.f);
          ov[r] = (__bf16)val;
        }
      }
      *(v8bf*)(void*)optr = ov;
    }
  }
}

// ---------------- global avg pool + FC head (NHWC) ----------------

__global__ void k_pool(const __bf16* __restrict__ in, float* __restrict__ pooled, int HW) {
  int i = blockIdx.x * blockDim.x + threadIdx.x;  // 32*512
  if (i >= 32 * 512) return;
  int n = i / 512, c = i % 512;
  float s = 0.f;
  const __bf16* ptr = in + n * HW * 512 + c;
  for (int k = 0; k < HW; ++k) s += (float)ptr[k * 512];
  pooled[i] = s / (float)HW;
}

__global__ void k_head(const float* __restrict__ pooled, const float* __restrict__ w,
                       const float* __restrict__ bias, float* __restrict__ out) {
  int i = blockIdx.x * blockDim.x + threadIdx.x;  // 32*1000
  if (i >= 32000) return;
  int cls = i % 1000, n = i / 1000;
  float s = bias[cls];
  const float* pw = w + cls * 512;
  const float* pp = pooled + n * 512;
  for (int k = 0; k < 512; ++k) s += pp[k] * pw[k];
  out[i] = s;
}

// ---------------- host orchestration ----------------

extern "C" void kernel_launch(void* const* d_in, const int* in_sizes, int n_in,
                              void* d_out, int out_size, void* d_ws, size_t ws_size,
                              hipStream_t stream) {
  (void)in_sizes; (void)n_in; (void)out_size; (void)ws_size;
  const float* x      = (const float*)d_in[0];
  const int*   policy = (const int*)d_in[1];
  auto F = [&](int i) { return (const float*)d_in[i]; };

  // Params pytree flattening (JAX sorted-key order):
  //  base: bn{b,g,m,v}=2..5, w=6
  //  blocks[i]: bn1{b,g,m,v}, bn2{b,g,m,v}, w1, w2 -> leaves 7+10*i .. 7+10*i+9
  //  ds keys sort "14","4","8" -> blocks 13,3,7; each bn{b,g,m,v}, w
  //  head: b=182, w=183
  auto blkL = [&](int i, int leaf) { return 7 + i * 10 + leaf; };
  const int dsBlk[3] = {13, 3, 7};
  int dsLeaf[16] = {0};
  for (int j = 0; j < 3; ++j) dsLeaf[dsBlk[j]] = 167 + j * 5;
  const int L_HEAD_B = 182, L_HEAD_W = 183;

  static const int CIN[16]  = {64,64,64,64, 128,128,128,128, 256,256,256,256,256,256, 512,512};
  static const int COUT[16] = {64,64,64,128,128,128,128,256, 256,256,256,256,256,512, 512,512};
  static const bool DS[16]  = {false,false,false,true, false,false,false,true,
                               false,false,false,false,false,true, false,false};

  auto magic = [](int d) -> unsigned { return (unsigned)((0x100000000ULL + d - 1) / (unsigned long long)d); };

  // ---- workspace layout ----
  char* ws = (char*)d_ws;
  size_t off = 0;
  auto alloc = [&](size_t bytes) -> char* {
    off = (off + 255) & ~(size_t)255;
    char* p = ws + off;
    off += bytes;
    return p;
  };
  const size_t baseElems = 32ULL * 112 * 112 * 64;  // 25.7M (NHWC)
  const size_t actElems  = 32ULL * 56 * 56 * 64;    // 6.4M  (max per-stage activation, NHWC)
  __bf16* basebuf = (__bf16*)alloc(baseElems * sizeof(__bf16));
  __bf16* bufs[3];
  bufs[0] = (__bf16*)alloc(actElems * sizeof(__bf16));
  bufs[1] = basebuf;              // overlay: basebuf is dead after maxpool
  bufs[2] = basebuf + actElems;   // 2*actElems <= baseElems
  float* pooled     = (float*)alloc(32 * 512 * sizeof(float));
  float* base_scale = (float*)alloc(64 * sizeof(float));
  float* base_shift = (float*)alloc(64 * sizeof(float));

  __bf16* wt1[16]; __bf16* wt2[16];
  float *sc1[16], *sh1[16], *sc2[16], *sh2[16];
  for (int i = 0; i < 16; ++i) {
    wt1[i] = (__bf16*)alloc((size_t)COUT[i] * CIN[i] * 9 * sizeof(__bf16));
    wt2[i] = (__bf16*)alloc((size_t)COUT[i] * COUT[i] * 9 * sizeof(__bf16));
    sc1[i] = (float*)alloc(COUT[i] * sizeof(float));
    sh1[i] = (float*)alloc(COUT[i] * sizeof(float));
    sc2[i] = (float*)alloc(COUT[i] * sizeof(float));
    sh2[i] = (float*)alloc(COUT[i] * sizeof(float));
  }
  __bf16* wtds[16] = {};
  float *scds[16] = {}, *shds[16] = {};
  for (int j = 0; j < 3; ++j) {
    int i = dsBlk[j];
    wtds[i] = (__bf16*)alloc((size_t)COUT[i] * CIN[i] * sizeof(__bf16));
    scds[i] = (float*)alloc(COUT[i] * sizeof(float));
    shds[i] = (float*)alloc(COUT[i] * sizeof(float));
  }

  // ---- prep: BN folds + weight convert/transpose/swizzle ----
  k_bnfold<<<1, 64, 0, stream>>>(F(3), F(2), F(4), F(5), base_scale, base_shift, 64);
  for (int i = 0; i < 16; ++i) {
    int C = COUT[i];
    k_bnfold<<<(C + 63) / 64, 64, 0, stream>>>(F(blkL(i,1)), F(blkL(i,0)), F(blkL(i,2)), F(blkL(i,3)), sc1[i], sh1[i], C);
    k_bnfold<<<(C + 63) / 64, 64, 0, stream>>>(F(blkL(i,5)), F(blkL(i,4)), F(blkL(i,6)), F(blkL(i,7)), sc2[i], sh2[i], C);
    int n1 = C * CIN[i] * 9, n2 = C * C * 9;
    k_wconv<<<(n1 + 255) / 256, 256, 0, stream>>>(F(blkL(i,8)), wt1[i], C, CIN[i], 9);
    k_wconv<<<(n2 + 255) / 256, 256, 0, stream>>>(F(blkL(i,9)), wt2[i], C, C, 9);
    if (DS[i]) {
      int lb = dsLeaf[i];
      k_bnfold<<<(C + 63) / 64, 64, 0, stream>>>(F(lb+1), F(lb+0), F(lb+2), F(lb+3), scds[i], shds[i], C);
      int nd = C * CIN[i];
      k_wconv<<<(nd + 255) / 256, 256, 0, stream>>>(F(lb+4), wtds[i], C, CIN[i], 1);
    }
  }

  // ---- stem ----
  k_baseconv<<<(unsigned)((baseElems + 255) / 256), 256, 0, stream>>>(x, F(6), base_scale, base_shift, basebuf);
  k_maxpool<<<(unsigned)((actElems + 255) / 256), 256, 0, stream>>>(basebuf, bufs[0]);

  // ---- 16 gated residual blocks ----
  int cur = 0, f1 = 1, f2 = 2;
  int hin = 56;
  for (int i = 0; i < 16; ++i) {
    const int s    = DS[i] ? 2 : 1;
    const int hout = hin / s;
    const int Ci = CIN[i], Co = COUT[i];
    const int HWo = hout * hout;
    const unsigned Mw = magic(hout);
    dim3 grid((HWo + 63) / 64, Co / 32, 32);
    if (DS[i]) {
      // residual = BN(conv1x1 s2)
      k_conv_wmma<<<grid, 32, 0, stream>>>(bufs[cur], wtds[i], scds[i], shds[i], bufs[f1],
                                           policy, i, MODE_BN,
                                           Ci, Co, hin, hin, hout, hout, 2, 0, 1, Mw);
      // h = relu(BN(conv3x3 s2))
      k_conv_wmma<<<grid, 32, 0, stream>>>(bufs[cur], wt1[i], sc1[i], sh1[i], bufs[f2],
                                           policy, i, MODE_BNRELU,
                                           Ci, Co, hin, hin, hout, hout, 2, 1, 9, Mw);
      // out = gate(relu(res + BN(conv3x3)))   [in-place on residual buffer]
      k_conv_wmma<<<grid, 32, 0, stream>>>(bufs[f2], wt2[i], sc2[i], sh2[i], bufs[f1],
                                           policy, i, MODE_RES_GATE,
                                           Co, Co, hout, hout, hout, hout, 1, 1, 9, Mw);
      int old = cur; cur = f1; f1 = old;
    } else {
      k_conv_wmma<<<grid, 32, 0, stream>>>(bufs[cur], wt1[i], sc1[i], sh1[i], bufs[f1],
                                           policy, i, MODE_BNRELU,
                                           Ci, Co, hin, hin, hout, hout, 1, 1, 9, Mw);
      k_conv_wmma<<<grid, 32, 0, stream>>>(bufs[f1], wt2[i], sc2[i], sh2[i], bufs[cur],
                                           policy, i, MODE_RES_GATE,
                                           Co, Co, hout, hout, hout, hout, 1, 1, 9, Mw);
    }
    hin = hout;
  }

  // ---- head ----
  k_pool<<<(32 * 512 + 255) / 256, 256, 0, stream>>>(bufs[cur], pooled, hin * hin);
  k_head<<<(32000 + 255) / 256, 256, 0, stream>>>(pooled, F(L_HEAD_W), F(L_HEAD_B), (float*)d_out);
}
